// NeuroSAT_5806795784638
// MI455X (gfx1250) — matrix operand order repaired and checked by reading, not compiled, not run on hardware
//
#include <hip/hip_runtime.h>
#include <hip/hip_bf16.h>
#include <math.h>

// ---------------------------------------------------------------------------
// NeuroSAT message passing on MI455X (gfx1250).
// All dense GEMMs use v_wmma_f32_16x16x32_bf16 (bf16 in, f32 accumulate).
// Block = 256 threads (8 waves) computing a 64x128 output tile:
//   - 64 x K A-tile staged once in LDS, shared by all waves
//   - each wave: one 16-col slice, 4 row-subtiles -> 4 accumulators,
//     so each B fragment (global, L2-resident weights) feeds 4 WMMAs.
// ---------------------------------------------------------------------------

#define N_VARS    4000
#define N_LITS    8000
#define N_CLAUSES 16000
#define N_CELLS   48000
#define DIM       128
#define N_ROUNDS  16
#define N_PROBS   8

typedef __attribute__((ext_vector_type(4)))  unsigned int   u32x4;
typedef __attribute__((ext_vector_type(4)))  float          f32x4;
typedef __attribute__((ext_vector_type(8)))  float          v8f;
typedef __attribute__((ext_vector_type(16))) __bf16         v16bf;

union Frag16 { v16bf v; u32x4 u[2]; };

__device__ __forceinline__ unsigned short f2bf(float x) {
    unsigned int u = __float_as_uint(x);
    u += 0x7FFFu + ((u >> 16) & 1u);          // round-to-nearest-even
    return (unsigned short)(u >> 16);
}

// ---------------------------------------------------------------------------
// Generic bf16 GEMM:  Cout[M,N] = act( A[M,K] * W[N,K]^T + bias + (Cin?) )
//   A row-major bf16 (stride K), W row-major bf16 [N,K] (== torch weight).
//   blockIdx.x -> 64-row tile, blockIdx.y -> 128-wide N slice.
//   M%64==0, N%128==0, K%32==0, K<=256 assumed (true for all call sites).
// ---------------------------------------------------------------------------
__global__ __launch_bounds__(256)
void k_gemm_bf16(const unsigned short* __restrict__ A,
                 const unsigned short* __restrict__ W,
                 const float* __restrict__ bias,
                 const float* Cin,            // nullable: accumulate source
                 float* Cout,
                 unsigned short* Obf,         // nullable: bf16 mirror of output
                 int M, int N, int K, int relu)
{
    __shared__ __align__(16) unsigned short As[64 * 264];   // K<=256, pad 8
    const int tid = threadIdx.x;
    const int m0  = blockIdx.x * 64;
    const int ldA = K + 8;

    // cooperative stage of the 64xK A tile into LDS (16B chunks)
    const int rowChunks = K >> 3;             // 8 bf16 per 16B chunk
    const int total     = rowChunks << 6;     // 64 rows
    for (int t = tid; t < total; t += 256) {
        int row = t / rowChunks;
        int c   = t - row * rowChunks;
        u32x4 v = *(const u32x4*)(A + (size_t)(m0 + row) * K + c * 8);
        *(u32x4*)&As[row * ldA + c * 8] = v;
    }
    __syncthreads();

    const int wave = tid >> 5;
    const int lane = tid & 31;
    const int l16  = lane & 15;               // output col within tile / A row
    const int half = lane >> 4;
    const int n    = blockIdx.y * 128 + wave * 16 + l16;

    v8f acc[4];
    const float b0 = bias ? bias[n] : 0.f;
    if (Cin) {
#pragma unroll
        for (int s = 0; s < 4; ++s)
#pragma unroll
            for (int r = 0; r < 8; ++r)
                acc[s][r] = Cin[(size_t)(m0 + s * 16 + r + 8 * half) * N + n] + b0;
    } else {
#pragma unroll
        for (int s = 0; s < 4; ++s)
#pragma unroll
            for (int r = 0; r < 8; ++r) acc[s][r] = b0;
    }

    const unsigned short* wrow = W + (size_t)n * K;
    for (int k0 = 0; k0 < K; k0 += 32) {
        // B fragment (32x16 layout): lane holds col l16, K = k0+half*16+[0..15]
        Frag16 b;
        const u32x4* wp = (const u32x4*)(wrow + k0 + half * 16);
        b.u[0] = wp[0];
        b.u[1] = wp[1];
#pragma unroll
        for (int s = 0; s < 4; ++s) {
            // A fragment (16x32 layout): lane holds row l16,
            // K = k0 + half*16 + [0..15]
            Frag16 a;
            const unsigned short* ar = &As[(s * 16 + l16) * ldA + k0 + half * 16];
            a.u[0] = *(const u32x4*)ar;
            a.u[1] = *(const u32x4*)(ar + 8);
            acc[s] = __builtin_amdgcn_wmma_f32_16x16x32_bf16(
                         false, a.v, false, b.v, (short)0, acc[s], false, false);
        }
    }

#pragma unroll
    for (int s = 0; s < 4; ++s)
#pragma unroll
        for (int r = 0; r < 8; ++r) {
            float v = acc[s][r];
            if (relu) v = fmaxf(v, 0.f);
            size_t o = (size_t)(m0 + s * 16 + r + 8 * half) * N + n;
            Cout[o] = v;
            if (Obf) Obf[o] = f2bf(v);
        }
}

// ---------------------------------------------------------------------------
// Sparse incidence scatter:  dst[drow] += src[srow]  per cell (128-dim rows).
// 32 lanes per cell, 4 floats per lane, fp32 global atomics.
// ---------------------------------------------------------------------------
__global__ __launch_bounds__(256)
void k_scatter_add(const int* __restrict__ idx, const float* __restrict__ src,
                   float* dst, int ncells, int to_clause)
{
    int t = blockIdx.x * 256 + threadIdx.x;
    if (t >= ncells * 32) return;
    int cell = t >> 5, lane = t & 31;
    int a = idx[cell * 2 + 0];                // literal
    int b = idx[cell * 2 + 1];                // clause
    int srow = to_clause ? a : b;
    int drow = to_clause ? b : a;
    f32x4 v = *(const f32x4*)(src + (size_t)srow * DIM + lane * 4);
    float* d = dst + (size_t)drow * DIM + lane * 4;
#pragma unroll
    for (int k = 0; k < 4; ++k)
        (void)__hip_atomic_fetch_add(&d[k], v[k], __ATOMIC_RELAXED,
                                     __HIP_MEMORY_SCOPE_AGENT);
}

__global__ void k_zero_f32(float* p, int n) {
    int t = blockIdx.x * 256 + threadIdx.x;
    if (t < n) p[t] = 0.f;
}

__global__ void k_f32_to_bf16(const float* __restrict__ s, unsigned short* d, int n) {
    int t = blockIdx.x * 256 + threadIdx.x;
    if (t < n) d[t] = f2bf(s[t]);
}

// h/c init:  h = broadcast(w[:,0] + b), c = 0
__global__ void k_init_state(const float* __restrict__ w, const float* __restrict__ b,
                             unsigned short* hbf, float* c, int rows)
{
    int t = blockIdx.x * 256 + threadIdx.x;
    if (t >= rows * DIM) return;
    int j = t & (DIM - 1);
    float v = w[j] + b[j];
    hbf[t] = f2bf(v);
    c[t]   = 0.f;
}

// LSTM elementwise (PyTorch gate order i,f,g,o; gates stride 512)
__global__ void k_lstm(const float* __restrict__ gates, float* c,
                       unsigned short* hbf, int rows)
{
    int t = blockIdx.x * 256 + threadIdx.x;
    if (t >= rows * DIM) return;
    int r = t >> 7, j = t & 127;
    const float* g = gates + (size_t)r * 512;
    float si = 1.f / (1.f + expf(-g[j]));
    float sf = 1.f / (1.f + expf(-g[128 + j]));
    float gg = g[256 + j];
    float so = 1.f / (1.f + expf(-g[384 + j]));
    float cn = sf * c[t] + si * tanhf(gg);
    c[t]   = cn;
    hbf[t] = f2bf(so * tanhf(cn));
}

// X[l] = [ bf16(CL_msg[l]) , Lh_bf[(l+N_VARS)%N_LITS] ]   (8000 x 256)
__global__ void k_build_x(const float* __restrict__ cl,
                          const unsigned short* __restrict__ lhbf,
                          unsigned short* X)
{
    int t = blockIdx.x * 256 + threadIdx.x;
    if (t >= N_LITS * DIM) return;
    int l = t >> 7, j = t & 127;
    int fl = (l < N_VARS) ? (l + N_VARS) : (l - N_VARS);
    X[(size_t)l * 256 + j]       = f2bf(cl[t]);
    X[(size_t)l * 256 + 128 + j] = lhbf[(size_t)fl * DIM + j];
}

// vote[l] = dot(v2[l,:], w3) + b3   — one wave32 per row
__global__ __launch_bounds__(256)
void k_vote(const float* __restrict__ v2, const float* __restrict__ w3,
            const float* __restrict__ b3, float* vote, int rows)
{
    int gt   = blockIdx.x * 256 + threadIdx.x;
    int row  = gt >> 5;
    int lane = gt & 31;
    if (row >= rows) return;
    const float* r = v2 + (size_t)row * DIM + lane * 4;
    const float* w = w3 + lane * 4;
    float s = r[0] * w[0] + r[1] * w[1] + r[2] * w[2] + r[3] * w[3];
    for (int m = 16; m; m >>= 1) s += __shfl_xor(s, m, 32);
    if (lane == 0) vote[row] = s + b3[0];
}

// out[p] = mean over v in [500p,500p+500) of (vote[v] + vote[N_VARS+v])
__global__ void k_reduce(const float* __restrict__ vote, float* out)
{
    __shared__ float sh[256];
    int p = blockIdx.x, t = threadIdx.x;
    float s = 0.f;
    for (int i = t; i < 500; i += 256)
        s += vote[500 * p + i] + vote[N_VARS + 500 * p + i];
    sh[t] = s;
    __syncthreads();
    for (int k = 128; k; k >>= 1) { if (t < k) sh[t] += sh[t + k]; __syncthreads(); }
    if (t == 0) out[p] = sh[0] * (1.0f / 1000.0f);
}

// ---------------------------------------------------------------------------

extern "C" void kernel_launch(void* const* d_in, const int* in_sizes, int n_in,
                              void* d_out, int out_size, void* d_ws, size_t ws_size,
                              hipStream_t stream)
{
    (void)in_sizes; (void)n_in; (void)out_size; (void)ws_size;

    const int*   idx      = (const int*)  d_in[0];
    const float* L_init_w = (const float*)d_in[4];
    const float* L_init_b = (const float*)d_in[5];
    const float* C_init_w = (const float*)d_in[6];
    const float* C_init_b = (const float*)d_in[7];
    const float* Lmsg_w[3] = {(const float*)d_in[8],  (const float*)d_in[10], (const float*)d_in[12]};
    const float* Lmsg_b[3] = {(const float*)d_in[9],  (const float*)d_in[11], (const float*)d_in[13]};
    const float* Cmsg_w[3] = {(const float*)d_in[14], (const float*)d_in[16], (const float*)d_in[18]};
    const float* Cmsg_b[3] = {(const float*)d_in[15], (const float*)d_in[17], (const float*)d_in[19]};
    const float* Lv_w1 = (const float*)d_in[20]; const float* Lv_b1 = (const float*)d_in[21];
    const float* Lv_w2 = (const float*)d_in[22]; const float* Lv_b2 = (const float*)d_in[23];
    const float* Lv_w3 = (const float*)d_in[24]; const float* Lv_b3 = (const float*)d_in[25];
    const float* Lu_wih = (const float*)d_in[26]; const float* Lu_whh = (const float*)d_in[27];
    const float* Lu_bih = (const float*)d_in[28]; const float* Lu_bhh = (const float*)d_in[29];
    const float* Cu_wih = (const float*)d_in[30]; const float* Cu_whh = (const float*)d_in[31];
    const float* Cu_bih = (const float*)d_in[32]; const float* Cu_bhh = (const float*)d_in[33];
    float* out = (float*)d_out;

    // ---- workspace carve-up -------------------------------------------------
    char* ws = (char*)d_ws;
    size_t off = 0;
    auto take = [&](size_t bytes) -> char* {
        char* p = ws + off;
        off = (off + bytes + 255) & ~(size_t)255;
        return p;
    };
    float*          Lc      = (float*)         take((size_t)N_LITS    * DIM * 4);
    float*          Cc      = (float*)         take((size_t)N_CLAUSES * DIM * 4);
    unsigned short* Lh_bf   = (unsigned short*)take((size_t)N_LITS    * DIM * 2);
    unsigned short* Ch_bf   = (unsigned short*)take((size_t)N_CLAUSES * DIM * 2);
    float*          tmp1    = (float*)         take((size_t)N_CLAUSES * DIM * 4);
    unsigned short* tmp1_bf = (unsigned short*)take((size_t)N_CLAUSES * DIM * 2);
    float*          tmp2    = (float*)         take((size_t)N_CLAUSES * DIM * 4);
    unsigned short* tmp2_bf = (unsigned short*)take((size_t)N_CLAUSES * DIM * 2);
    float*          msg     = (float*)         take((size_t)N_CLAUSES * DIM * 4);
    unsigned short* msg_bf  = (unsigned short*)take((size_t)N_CLAUSES * DIM * 2);
    unsigned short* X_bf    = (unsigned short*)take((size_t)N_LITS    * 256 * 2);
    float*          gates   = (float*)         take((size_t)N_CLAUSES * 512 * 4);
    float*          vote    = (float*)         take((size_t)N_LITS * 4);
    unsigned short* wLm[3], *wCm[3];
    for (int i = 0; i < 3; ++i) wLm[i] = (unsigned short*)take(DIM * DIM * 2);
    for (int i = 0; i < 3; ++i) wCm[i] = (unsigned short*)take(DIM * DIM * 2);
    unsigned short* wLv1   = (unsigned short*)take(DIM * DIM * 2);
    unsigned short* wLv2   = (unsigned short*)take(DIM * DIM * 2);
    unsigned short* wLuIH  = (unsigned short*)take(512 * 256 * 2);
    unsigned short* wLuHH  = (unsigned short*)take(512 * 128 * 2);
    unsigned short* wCuIH  = (unsigned short*)take(512 * 128 * 2);
    unsigned short* wCuHH  = (unsigned short*)take(512 * 128 * 2);

    auto conv = [&](const float* s, unsigned short* d, int n) {
        k_f32_to_bf16<<<(n + 255) / 256, 256, 0, stream>>>(s, d, n);
    };
    auto gemm = [&](const unsigned short* A, const unsigned short* W,
                    const float* bias, const float* Cin, float* Cout,
                    unsigned short* Obf, int M, int N, int K, int relu) {
        dim3 grid(M / 64, N / 128);
        k_gemm_bf16<<<grid, 256, 0, stream>>>(A, W, bias, Cin, Cout, Obf, M, N, K, relu);
    };

    // ---- weight conversion (every call; deterministic) ----------------------
    for (int i = 0; i < 3; ++i) conv(Lmsg_w[i], wLm[i], DIM * DIM);
    for (int i = 0; i < 3; ++i) conv(Cmsg_w[i], wCm[i], DIM * DIM);
    conv(Lv_w1, wLv1, DIM * DIM);
    conv(Lv_w2, wLv2, DIM * DIM);
    conv(Lu_wih, wLuIH, 512 * 256);
    conv(Lu_whh, wLuHH, 512 * 128);
    conv(Cu_wih, wCuIH, 512 * 128);
    conv(Cu_whh, wCuHH, 512 * 128);

    // ---- state init ---------------------------------------------------------
    k_init_state<<<(N_LITS * DIM + 255) / 256, 256, 0, stream>>>(L_init_w, L_init_b, Lh_bf, Lc, N_LITS);
    k_init_state<<<(N_CLAUSES * DIM + 255) / 256, 256, 0, stream>>>(C_init_w, C_init_b, Ch_bf, Cc, N_CLAUSES);

    // ---- 16 message-passing rounds -----------------------------------------
    for (int r = 0; r < N_ROUNDS; ++r) {
        // L_pre = MLP(Lh)
        gemm(Lh_bf,   wLm[0], Lmsg_b[0], nullptr, tmp1, tmp1_bf, N_LITS, DIM, DIM, 1);
        gemm(tmp1_bf, wLm[1], Lmsg_b[1], nullptr, tmp2, tmp2_bf, N_LITS, DIM, DIM, 1);
        gemm(tmp2_bf, wLm[2], Lmsg_b[2], nullptr, tmp1, nullptr, N_LITS, DIM, DIM, 0);
        // LC_msg = M^T @ L_pre
        k_zero_f32<<<(N_CLAUSES * DIM + 255) / 256, 256, 0, stream>>>(msg, N_CLAUSES * DIM);
        k_scatter_add<<<(N_CELLS * 32 + 255) / 256, 256, 0, stream>>>(idx, tmp1, msg, N_CELLS, 1);
        conv(msg, msg_bf, N_CLAUSES * DIM);
        // C-LSTM:  gates = LC@Wih^T + bih + Ch@Whh^T + bhh
        gemm(msg_bf, wCuIH, Cu_bih, nullptr, gates, nullptr, N_CLAUSES, 512, DIM, 0);
        gemm(Ch_bf,  wCuHH, Cu_bhh, gates,   gates, nullptr, N_CLAUSES, 512, DIM, 0);
        k_lstm<<<(N_CLAUSES * DIM + 255) / 256, 256, 0, stream>>>(gates, Cc, Ch_bf, N_CLAUSES);
        // C_pre = MLP(Ch)
        gemm(Ch_bf,   wCm[0], Cmsg_b[0], nullptr, tmp1, tmp1_bf, N_CLAUSES, DIM, DIM, 1);
        gemm(tmp1_bf, wCm[1], Cmsg_b[1], nullptr, tmp2, tmp2_bf, N_CLAUSES, DIM, DIM, 1);
        gemm(tmp2_bf, wCm[2], Cmsg_b[2], nullptr, tmp1, nullptr, N_CLAUSES, DIM, DIM, 0);
        // CL_msg = M @ C_pre
        k_zero_f32<<<(N_LITS * DIM + 255) / 256, 256, 0, stream>>>(msg, N_LITS * DIM);
        k_scatter_add<<<(N_CELLS * 32 + 255) / 256, 256, 0, stream>>>(idx, tmp1, msg, N_CELLS, 0);
        // X = [CL_msg, flip(Lh)] ; L-LSTM
        k_build_x<<<(N_LITS * DIM + 255) / 256, 256, 0, stream>>>(msg, Lh_bf, X_bf);
        gemm(X_bf,  wLuIH, Lu_bih, nullptr, gates, nullptr, N_LITS, 512, 256, 0);
        gemm(Lh_bf, wLuHH, Lu_bhh, gates,   gates, nullptr, N_LITS, 512, DIM, 0);
        k_lstm<<<(N_LITS * DIM + 255) / 256, 256, 0, stream>>>(gates, Lc, Lh_bf, N_LITS);
    }

    // ---- vote head + problem means -----------------------------------------
    gemm(Lh_bf,   wLv1, Lv_b1, nullptr, tmp1, tmp1_bf, N_LITS, DIM, DIM, 1);
    gemm(tmp1_bf, wLv2, Lv_b2, nullptr, tmp2, tmp2_bf, N_LITS, DIM, DIM, 1);
    k_vote<<<(N_LITS * 32 + 255) / 256, 256, 0, stream>>>(tmp2, Lv_w3, Lv_b3, vote, N_LITS);
    k_reduce<<<N_PROBS, 256, 0, stream>>>(vote, out);
}
